// LatConv_86861418595019
// MI455X (gfx1250) — matrix operand order repaired and checked by reading, not compiled, not run on hardware
//
#include <hip/hip_runtime.h>

// Reference shapes: B=4, C=64, H=256, W=512, k=3
#define C_CH    64
#define HW      131072          // 256*512
#define KW      1536            // 3*512 (gathered image width)
#define KDIM    576             // C*9, K = ci*9 + t (flat weight row layout)
#define NTILE   32              // spatial positions per block (2 x 16 per wave)
#define THREADS 128             // 4 waves: each wave owns one 16-co stripe
#define WELEMS  (C_CH * KDIM)   // 36864 weight elements

typedef __attribute__((ext_vector_type(16))) __bf16 v16bf;
typedef __attribute__((ext_vector_type(8)))  float  v8f;

union Frag16 { __bf16 e[16]; v16bf v; uint4 u[2]; };

// ---- Prep: split f32 weights into bf16 hi/lo planes (block-invariant) ----
__global__ __launch_bounds__(256)
void prep_weights(const float* __restrict__ w,
                  __bf16* __restrict__ wh, __bf16* __restrict__ wl)
{
    const int i = blockIdx.x * 256 + threadIdx.x;
    if (i < WELEMS) {
        const float f = w[i];
        const __bf16 h = (__bf16)f;
        wh[i] = h;
        wl[i] = (__bf16)(f - (float)h);
    }
}

// ---- Main: gathered GEMM, 3-term bf16 split (Wh*Xh + Wh*Xl + Wl*Xh) ----
__global__ __launch_bounds__(THREADS)
void latconv_wmma_bf16x3(const float* __restrict__ x,
                         const long long* __restrict__ idx_flat,
                         const __bf16* __restrict__ Wh,
                         const __bf16* __restrict__ Wl,
                         const float* __restrict__ bias,
                         float* __restrict__ out)
{
    __shared__ int pidx[9 * NTILE];                      // 1152 B
    __shared__ __align__(32) __bf16 XhT[NTILE * KDIM];   // 36 KB, [n][K]
    __shared__ __align__(32) __bf16 XlT[NTILE * KDIM];   // 36 KB

    const int tid = threadIdx.x;
    const int b   = blockIdx.x >> 12;            // 4096 tiles per batch
    const int s0  = (blockIdx.x & 4095) << 5;    // 32 positions, single oh row
    const int oh  = s0 >> 9;
    const int ow0 = s0 & 511;

    // ---- Phase 1: resolve the 9*32 gather indices (division-free) ----
    for (int e = tid; e < 9 * NTILE; e += THREADS) {
        const int t  = e >> 5;                   // kh*3+kw, 0..8
        const int n  = e & 31;
        const int kh = (t * 11) >> 5;            // exact t/3 for t in [0,8]
        const int kw = t - 3 * kh;
        const int r  = 3 * oh + kh;
        const int c  = 3 * (ow0 + n) + kw;
        pidx[e] = (int)idx_flat[(long long)r * KW + c];
    }
    __syncthreads();

    // ---- Phase 2: gather x into LDS, bf16 hi/lo split (division-free) ----
    {
        const int n  = tid >> 2;                 // 0..31 spatial position
        const int l4 = tid & 3;                  // 4 threads per position
        int p[9];
#pragma unroll
        for (int t = 0; t < 9; ++t) p[t] = pidx[t * NTILE + n];

        const float* xb = x + (size_t)b * (C_CH * (size_t)HW);
        __bf16* xh_row = XhT + n * KDIM;
        __bf16* xl_row = XlT + n * KDIM;
        for (int ci = l4; ci < C_CH; ci += 4) {
            const float* xcol = xb + (size_t)ci * HW;
            const int Kb = ci * 9;
#pragma unroll
            for (int t = 0; t < 9; ++t) {
                const float f = xcol[p[t]];
                const __bf16 hi = (__bf16)f;
                xh_row[Kb + t] = hi;
                xl_row[Kb + t] = (__bf16)(f - (float)hi);
            }
        }
    }
    __syncthreads();

    // ---- Phase 3: per-wave 16co x 32n WMMA GEMM over K=576 ----
    const int lane   = tid & 31;
    const int wave   = tid >> 5;
    const int coBase = wave << 4;
    const int hiHalf = lane >> 4;
    const int mrow   = lane & 15;

    // D/C layout: VGPR j -> M = j + 8*hiHalf, N = lane&15. Bias into C.
    v8f acc0, acc1;
#pragma unroll
    for (int j = 0; j < 8; ++j) {
        const float bv = bias[coBase + j + 8 * hiHalf];
        acc0[j] = bv;
        acc1[j] = bv;
    }

    const __bf16* whrow  = Wh + (size_t)(coBase + mrow) * KDIM;
    const __bf16* wlrow  = Wl + (size_t)(coBase + mrow) * KDIM;
    const __bf16* bhrow0 = XhT + mrow * KDIM;          // n-tile 0: n = 0..15
    const __bf16* blrow0 = XlT + mrow * KDIM;
    const __bf16* bhrow1 = XhT + (16 + mrow) * KDIM;   // n-tile 1: n = 16..31
    const __bf16* blrow1 = XlT + (16 + mrow) * KDIM;
    // A k order used here (contiguous, same as B): lanes<16 hold K 0..15; lanes>=16 hold K 16..31
    const int aoff = hiHalf * 16;
    // B k order used here (contiguous, same as A): lanes<16 hold K 0..15; lanes>=16 hold K 16..31
    const int boff = hiHalf * 16;

#pragma unroll
    for (int kk = 0; kk < KDIM / 32; ++kk) {
        const int K0 = kk * 32;

        Frag16 ah, al, bh0, bl0, bh1, bl1;
        // A fragments: two aligned 16B runs each -> global_load_b128, shared
        // by both n-tiles (2x A reuse vs single-tile version)
        ah.u[0] = *(const uint4*)(whrow + K0 + aoff);
        ah.u[1] = *(const uint4*)(whrow + K0 + 8 + aoff);
        al.u[0] = *(const uint4*)(wlrow + K0 + aoff);
        al.u[1] = *(const uint4*)(wlrow + K0 + 8 + aoff);
        // B fragments: contiguous 32B LDS loads
        bh0.v = *(const v16bf*)__builtin_assume_aligned(bhrow0 + K0 + boff, 32);
        bl0.v = *(const v16bf*)__builtin_assume_aligned(blrow0 + K0 + boff, 32);
        bh1.v = *(const v16bf*)__builtin_assume_aligned(bhrow1 + K0 + boff, 32);
        bl1.v = *(const v16bf*)__builtin_assume_aligned(blrow1 + K0 + boff, 32);

        // Interleave the two independent accumulator chains for ILP.
        acc0 = __builtin_amdgcn_wmma_f32_16x16x32_bf16(false, ah.v, false, bh0.v,
                                                       (short)0, acc0, false, false);
        acc1 = __builtin_amdgcn_wmma_f32_16x16x32_bf16(false, ah.v, false, bh1.v,
                                                       (short)0, acc1, false, false);
        acc0 = __builtin_amdgcn_wmma_f32_16x16x32_bf16(false, ah.v, false, bl0.v,
                                                       (short)0, acc0, false, false);
        acc1 = __builtin_amdgcn_wmma_f32_16x16x32_bf16(false, ah.v, false, bl1.v,
                                                       (short)0, acc1, false, false);
        acc0 = __builtin_amdgcn_wmma_f32_16x16x32_bf16(false, al.v, false, bh0.v,
                                                       (short)0, acc0, false, false);
        acc1 = __builtin_amdgcn_wmma_f32_16x16x32_bf16(false, al.v, false, bh1.v,
                                                       (short)0, acc1, false, false);
    }

    // ---- Store: out[(b*64+co)*HW + s], 16-wide coalesced per VGPR row ----
    float* obase = out + (size_t)b * (C_CH * (size_t)HW) + s0 + mrow;
#pragma unroll
    for (int j = 0; j < 8; ++j) {
        const int co = coBase + j + 8 * hiHalf;
        obase[(size_t)co * HW]      = acc0[j];
        obase[(size_t)co * HW + 16] = acc1[j];
    }
}

extern "C" void kernel_launch(void* const* d_in, const int* in_sizes, int n_in,
                              void* d_out, int out_size, void* d_ws, size_t ws_size,
                              hipStream_t stream) {
    (void)in_sizes; (void)n_in; (void)ws_size; (void)out_size;
    const float*     xx   = (const float*)d_in[0];
    const long long* idx  = (const long long*)d_in[1];   // int64 per reference
    const float*     wgt  = (const float*)d_in[2];
    const float*     bias = (const float*)d_in[3];
    float*           out  = (float*)d_out;

    // Workspace: bf16 hi/lo weight planes (2 * 36864 * 2B = 147456 B)
    __bf16* wh = (__bf16*)d_ws;
    __bf16* wl = wh + WELEMS;

    prep_weights<<<(WELEMS + 255) / 256, 256, 0, stream>>>(wgt, wh, wl);

    const int grid = 4 * (HW / NTILE);   // 16384 blocks: (b, 32-pos tile)
    latconv_wmma_bf16x3<<<grid, THREADS, 0, stream>>>(xx, idx, wh, wl, bias, out);
}
